// ResidualVectorQuantizer_86723979640970
// MI455X (gfx1250) — matrix operand order repaired and checked by reading, not compiled, not run on hardware
//
#include <hip/hip_runtime.h>

#define L_LAYERS 4
#define K_CODES  1024
#define D_DIM    256
#define B_BATCH  32768
#define M_TILE   64
#define RSTR     264   // padded residual row stride in halves (264*2=528B, 16B aligned)

typedef __attribute__((ext_vector_type(16))) _Float16 v16h;
typedef __attribute__((ext_vector_type(8)))  _Float16 v8h;
typedef __attribute__((ext_vector_type(8)))  float    v8f;

// ---------------- prep: codebooks f32 -> f16, plus per-code squared norms ----------------
__global__ void rvq_prep(const float* __restrict__ cb,
                         _Float16* __restrict__ cbh,
                         float* __restrict__ ccn) {
  const int row = blockIdx.x;          // 0 .. L*K-1
  const int tid = threadIdx.x;         // 0 .. 255 (one per dim)
  const size_t off = (size_t)row * D_DIM + tid;
  float v = cb[off];
  cbh[off] = (_Float16)v;
  float s = v * v;
  #pragma unroll
  for (int o = 16; o >= 1; o >>= 1) s += __shfl_xor(s, o, 32);
  __shared__ float part[8];
  if ((tid & 31) == 0) part[tid >> 5] = s;
  __syncthreads();
  if (tid == 0) {
    float t = 0.f;
    #pragma unroll
    for (int i = 0; i < 8; ++i) t += part[i];
    ccn[row] = t;
  }
}

// ---------------- main RVQ kernel: one 64-row batch tile per workgroup ----------------
__global__ void __launch_bounds__(256)
rvq_main(const float* __restrict__ z,
         const float* __restrict__ cb,       // original fp32 codebooks (for residual update)
         const _Float16* __restrict__ cbh,   // f16 codebooks (L2-resident)
         const float* __restrict__ ccn,      // per-code squared norms
         float* __restrict__ out_q,
         float* __restrict__ out_probs,
         float* __restrict__ out_ids) {
  extern __shared__ char smem[];
  float*    scores  = (float*)smem;                                   // [64][1024] f32
  _Float16* resid   = (_Float16*)(smem + M_TILE * K_CODES * 4);       // [64][RSTR] f16
  float*    rr      = (float*)(smem + M_TILE * K_CODES * 4 + M_TILE * RSTR * 2); // [64]
  float*    red_sum = rr + 64;               // [2][64]
  float*    red_min = red_sum + 128;         // [2][64]
  int*      red_idx = (int*)(red_min + 128); // [2][64]
  float*    inv_sum = (float*)(red_idx + 128); // [64]
  int*      fin_idx = (int*)(inv_sum + 64);    // [64]

  const int tid    = threadIdx.x;
  const int lane   = tid & 31;
  const int wid    = tid >> 5;
  const int strip  = wid & 3;     // which 16-row strip of the tile
  const int nhalf  = wid >> 2;    // which 512-column half of the codebook
  const int half   = lane >> 4;   // half-wave index for WMMA layouts
  const int nloc   = lane & 15;   // lane-within-half (M for A, N for B/C)
  const int base   = blockIdx.x * M_TILE;

  // residual := z (f16 in LDS)
  for (int i = 0; i < M_TILE; ++i)
    resid[i * RSTR + tid] = (_Float16)z[(size_t)(base + i) * D_DIM + tid];
  __syncthreads();

  for (int l = 0; l < L_LAYERS; ++l) {
    // ---- per-row ||residual||^2 (4 threads per row, vectorized LDS reads) ----
    {
      const int row = tid >> 2, part = tid & 3;
      const v8h* rrow = (const v8h*)(resid + row * RSTR + part * 64);
      float s = 0.f;
      #pragma unroll
      for (int c8 = 0; c8 < 8; ++c8) {
        v8h x = rrow[c8];
        #pragma unroll
        for (int j = 0; j < 8; ++j) { float f = (float)x[j]; s += f * f; }
      }
      s += __shfl_xor(s, 1, 32);
      s += __shfl_xor(s, 2, 32);
      if (part == 0) rr[row] = s;
    }
    __syncthreads();

    // ---- load A fragments once per layer (16x32 f16 layout, k-contiguous per half) ----
    v16h afr[8];
    {
      const _Float16* arow = resid + (strip * 16 + nloc) * RSTR;  // m = lane&15
      #pragma unroll
      for (int kf = 0; kf < 8; ++kf) {
        const int k0 = kf * 32;
        v8h lo = *(const v8h*)(arow + k0 + half * 16);
        v8h hi = *(const v8h*)(arow + k0 + half * 16 + 8);
        afr[kf] = __builtin_shufflevector(lo, hi,
            0,1,2,3,4,5,6,7,8,9,10,11,12,13,14,15);
      }
    }
    float rrv[8];
    #pragma unroll
    for (int v = 0; v < 8; ++v) rrv[v] = rr[strip * 16 + half * 8 + v];

    float sum_acc[8]; float min_acc[8]; int idx_acc[8];
    #pragma unroll
    for (int v = 0; v < 8; ++v) { sum_acc[v] = 0.f; min_acc[v] = 3.4e38f; idx_acc[v] = 0; }

    const _Float16* cbl = cbh + (size_t)l * K_CODES * D_DIM;
    const float*    ccl = ccn + l * K_CODES;
    // per-lane base: column (code) row for this lane, fragment-contiguous reads
    const _Float16* brow0 = cbl + (size_t)(nhalf * 512 + nloc) * D_DIM + half * 16;

    const int srow0 = (strip * 16 + half * 8) * 1024 + nloc;  // scores base for v=0

    // B-fragment loader for one 16-column tile (16 x b128 loads, all independent)
    auto loadB = [&](v16h* dst, int nt) {
      const _Float16* bp = brow0 + (size_t)nt * 16 * D_DIM;
      #pragma unroll
      for (int kf = 0; kf < 8; ++kf)
        dst[kf] = *(const v16h*)(bp + kf * 32);
    };

    // epilogue: distances -> unnormalized softmax into LDS + running sum/argmin
    auto epilogue = [&](const v8f& c, int nt) {
      const int n0 = nhalf * 512 + nt * 16;
      const float cc = ccl[n0 + nloc];
      #pragma unroll
      for (int v = 0; v < 8; ++v) {
        float d2   = rrv[v] + cc - 2.0f * c[v];
        float dist = __builtin_amdgcn_sqrtf(__builtin_fmaxf(d2, 0.f));
        float p    = __expf(-2.0f * dist);        // 1/TEMPERATURE = 2; dist>=0, no max-shift needed
        scores[srow0 + v * 1024 + n0] = p;
        sum_acc[v] += p;
        if (dist < min_acc[v]) { min_acc[v] = dist; idx_acc[v] = n0 + nloc; }
      }
    };

    // ---- GEMM over this wave's 32 N-tiles, double-buffered B fragments ----
    v16h b0[8], b1[8];
    loadB(b0, 0);
    for (int nt = 0; nt < 32; nt += 2) {
      loadB(b1, nt + 1);                       // in flight during tile nt compute
      {
        v8f c = {};
        #pragma unroll
        for (int kf = 0; kf < 8; ++kf)
          c = __builtin_amdgcn_wmma_f32_16x16x32_f16(
                  false, afr[kf], false, b0[kf], (short)0, c, false, false);
        epilogue(c, nt);
      }
      if (nt < 30) loadB(b0, nt + 2);          // in flight during tile nt+1 compute
      {
        v8f c = {};
        #pragma unroll
        for (int kf = 0; kf < 8; ++kf)
          c = __builtin_amdgcn_wmma_f32_16x16x32_f16(
                  false, afr[kf], false, b1[kf], (short)0, c, false, false);
        epilogue(c, nt + 1);
      }
    }

    // ---- reduce sum / argmin across the 16 lanes of each half-wave ----
    #pragma unroll
    for (int off = 1; off < 16; off <<= 1) {
      #pragma unroll
      for (int v = 0; v < 8; ++v) {
        sum_acc[v] += __shfl_xor(sum_acc[v], off, 32);
        float om = __shfl_xor(min_acc[v], off, 32);
        int   oi = __shfl_xor(idx_acc[v], off, 32);
        if (om < min_acc[v] || (om == min_acc[v] && oi < idx_acc[v])) {
          min_acc[v] = om; idx_acc[v] = oi;
        }
      }
    }
    if (nloc == 0) {
      #pragma unroll
      for (int v = 0; v < 8; ++v) {
        const int row = strip * 16 + half * 8 + v;
        red_sum[nhalf * 64 + row] = sum_acc[v];
        red_min[nhalf * 64 + row] = min_acc[v];
        red_idx[nhalf * 64 + row] = idx_acc[v];
      }
    }
    __syncthreads();

    // ---- finalize per-row softmax denom + argmin, emit ids ----
    if (tid < 64) {
      float s  = red_sum[tid] + red_sum[64 + tid];
      float m0 = red_min[tid], m1 = red_min[64 + tid];
      int   i0 = red_idx[tid], i1 = red_idx[64 + tid];
      int idx = (m1 < m0 || (m1 == m0 && i1 < i0)) ? i1 : i0;
      inv_sum[tid] = 1.0f / s;
      fin_idx[tid] = idx;
      out_ids[(size_t)l * B_BATCH + base + tid] = (float)idx;
    }
    __syncthreads();

    // ---- normalize probs from LDS and stream to HBM (single pass, float4) ----
    {
      float* pbase = out_probs + (size_t)l * B_BATCH * K_CODES;
      for (int m = 0; m < 16; ++m) {
        const int row = strip * 16 + m;
        const float inv = inv_sum[row];
        const float4* src = (const float4*)(scores + row * 1024 + nhalf * 512);
        float4* dst = (float4*)(pbase + (size_t)(base + row) * K_CODES + nhalf * 512);
        for (int i = lane; i < 128; i += 32) {
          float4 p = src[i];
          p.x *= inv; p.y *= inv; p.z *= inv; p.w *= inv;
          dst[i] = p;
        }
      }
    }
    // ---- residual -= selected fp32 codebook row ----
    {
      const float* cbf = cb + (size_t)l * K_CODES * D_DIM;
      for (int i = 0; i < M_TILE; ++i) {
        const float cv = cbf[(size_t)fin_idx[i] * D_DIM + tid];
        float r = (float)resid[i * RSTR + tid] - cv;
        resid[i * RSTR + tid] = (_Float16)r;
      }
    }
    __syncthreads();
  }

  // quantized = z - final residual
  for (int i = 0; i < M_TILE; ++i) {
    const size_t off = (size_t)(base + i) * D_DIM + tid;
    out_q[off] = z[off] - (float)resid[i * RSTR + tid];
  }
}

extern "C" void kernel_launch(void* const* d_in, const int* in_sizes, int n_in,
                              void* d_out, int out_size, void* d_ws, size_t ws_size,
                              hipStream_t stream) {
  (void)in_sizes; (void)n_in; (void)out_size; (void)ws_size;

  const float* z  = (const float*)d_in[0];                  // [32768, 256]
  const float* cb = (const float*)d_in[1];                  // [4, 1024, 256]

  _Float16* cbh = (_Float16*)d_ws;                          // 2 MB f16 codebooks
  float*    ccn = (float*)((char*)d_ws +
                  (size_t)L_LAYERS * K_CODES * D_DIM * sizeof(_Float16)); // 16 KB norms

  float* out      = (float*)d_out;
  float* out_q    = out;                                    // [32768, 256]
  float* out_prob = out + (size_t)B_BATCH * D_DIM;          // [4, 32768, 1024]
  float* out_ids  = out_prob + (size_t)L_LAYERS * B_BATCH * K_CODES; // [4, 32768]

  rvq_prep<<<L_LAYERS * K_CODES, 256, 0, stream>>>(cb, cbh, ccn);

  const int smem_bytes = M_TILE * K_CODES * 4      // scores
                       + M_TILE * RSTR * 2         // residual f16
                       + (64 + 128 + 128 + 128 + 64 + 64) * 4; // reductions
  static_assert(M_TILE * K_CODES * 4 + M_TILE * RSTR * 2 + 576 * 4 <= 320 * 1024,
                "LDS budget exceeded");
  (void)hipFuncSetAttribute((const void*)rvq_main,
                            hipFuncAttributeMaxDynamicSharedMemorySize, smem_bytes);

  rvq_main<<<B_BATCH / M_TILE, 256, smem_bytes, stream>>>(
      z, cb, cbh, ccn, out_q, out_prob, out_ids);
}
